// MultiHeadLatentAttentionReference_47433618817305
// MI455X (gfx1250) — matrix operand order repaired and checked by reading, not compiled, not run on hardware
//
#include <hip/hip_runtime.h>
#include <math.h>

typedef __bf16 bf16;
typedef __attribute__((ext_vector_type(16))) __bf16 v16bf;
typedef __attribute__((ext_vector_type(8)))  __bf16 v8bf;
typedef __attribute__((ext_vector_type(8)))  float  v8f;
typedef __attribute__((ext_vector_type(4)))  unsigned int u32x4;
typedef __attribute__((ext_vector_type(8)))  int i32x8;
typedef __attribute__((ext_vector_type(4)))  int i32x4;

#define BM 128
#define BN 128
#define BK 32
#define KPAD 40   // LDS row stride in bf16 elements (80B = 20 banks, 16B aligned)

#if __has_builtin(__builtin_amdgcn_tensor_load_to_lds)
#define USE_TDM 1
#else
#define USE_TDM 0
#endif

__device__ __forceinline__ v8f wmma_bf16(v16bf a, v16bf b, v8f c) {
    return __builtin_amdgcn_wmma_f32_16x16x32_bf16(false, a, false, b, (short)0, c, false, false);
}

__device__ __forceinline__ v16bf cat16(v8bf lo, v8bf hi) {
    return __builtin_shufflevector(lo, hi, 0,1,2,3,4,5,6,7,8,9,10,11,12,13,14,15);
}

// ---- per-lane async fallback: 16B global -> LDS (ASYNCcnt) ----
__device__ __forceinline__ void async_copy16(const bf16* g, bf16* l) {
    unsigned lds = (unsigned)(unsigned long long)l;
    asm volatile("global_load_async_to_lds_b128 %0, %1, off"
                 :: "v"(lds), "v"(g) : "memory");
}

template<int N>
__device__ __forceinline__ void wait_async() {
#if __has_builtin(__builtin_amdgcn_s_wait_asynccnt)
    __builtin_amdgcn_s_wait_asynccnt(N);
#else
    asm volatile("s_wait_asynccnt %0" :: "i"(N) : "memory");
#endif
}

#if USE_TDM
template<int N>
__device__ __forceinline__ void wait_tensor() {
#if __has_builtin(__builtin_amdgcn_s_wait_tensorcnt)
    __builtin_amdgcn_s_wait_tensorcnt(N);
#else
    asm volatile("s_wait_tensorcnt %0" :: "i"(N) : "memory");
#endif
}

// TDM 2D tile load: rows x row_elems bf16 tile at gaddr (row stride ld_elems)
// into LDS at lds_byte_addr with an 80B padded row stride:
//   pad_interval code 3 = 16 DWORDs (one 64B row), pad_amount code 3 = 4 DWORDs.
// D# per CDNA5 ISA ch.8: group0 {count,lds,global,type}, group1 {dims/strides}.
__device__ __forceinline__ void tdm_load_2d(const void* gaddr, unsigned lds_byte_addr,
                                            int rows, int row_elems, long long ld_elems)
{
    unsigned long long ga = (unsigned long long)gaddr;        // 57-bit byte addr
    unsigned long long stride = (unsigned long long)ld_elems; // in data_size units

    u32x4 g0;
    g0[0] = 1u;                                              // count=1 (valid user D#)
    g0[1] = lds_byte_addr;                                   // lds_addr [63:32]
    g0[2] = (unsigned)(ga & 0xFFFFFFFFu);                    // global_addr[31:0]
    g0[3] = (unsigned)((ga >> 32) & 0x01FFFFFFu) | (2u << 30); // ga[56:32] | type=2

    i32x8 g1;
    g1[0] = (1 << 16)            // data_size = 1 -> 2 bytes
          | (1 << 20)            // pad_enable
          | (3 << 22)            // pad_interval code 3 = 16 DWORDs (64B)
          | (3 << 25);           // pad_amount  code 3 = 4 DWORDs (16B)
    g1[1] = (row_elems & 0xFFFF) << 16;                      // tensor_dim0[15:0] @63:48
    g1[2] = ((unsigned)row_elems >> 16)                      // tensor_dim0[31:16]
          | ((rows & 0xFFFF) << 16);                         // tensor_dim1[15:0] @95:80
    g1[3] = ((unsigned)rows >> 16)                           // tensor_dim1[31:16]
          | ((row_elems & 0xFFFF) << 16);                    // tile_dim0 @127:112
    g1[4] = (rows & 0xFFFF);                                 // tile_dim1 (tile_dim2=0)
    g1[5] = (int)(unsigned)(stride & 0xFFFFFFFFu);           // tensor_dim0_stride[31:0]
    g1[6] = (int)(unsigned)((stride >> 32) & 0xFFFFu);       // stride[47:32]; d1_stride=0
    g1[7] = 0;

    i32x4 gz;  gz[0] = 0; gz[1] = 0; gz[2] = 0; gz[3] = 0;
#if defined(__clang_major__) && (__clang_major__ >= 23)
    i32x8 gz8; gz8[0]=0; gz8[1]=0; gz8[2]=0; gz8[3]=0; gz8[4]=0; gz8[5]=0; gz8[6]=0; gz8[7]=0;
    __builtin_amdgcn_tensor_load_to_lds(g0, g1, gz, gz, gz8, 0);
#else
    __builtin_amdgcn_tensor_load_to_lds(g0, g1, gz, gz, 0);
#endif
}
#endif // USE_TDM

// ---------------------------------------------------------------------------
// NT bf16 GEMM: C[M,N] = A[M,K] * B[N,K]^T, fp32 accumulate, double-buffered
// LDS staged by the Tensor Data Mover (TENSORcnt) or per-lane async copies.
// ---------------------------------------------------------------------------
template<bool OUT_BF16>
__global__ __launch_bounds__(256) void gemm_nt(
    const bf16* __restrict__ Ag, const bf16* __restrict__ Bg, void* __restrict__ Cg,
    int K, int lda, int ldb, int ldc,
    long long sAb, long long sAh, long long sBb, long long sBh,
    long long sCb, long long sCh, int H)
{
    const int z  = blockIdx.z;
    const int hh = z % H;
    const int bb = z / H;
    const bf16* A = Ag + bb * sAb + hh * sAh;
    const bf16* B = Bg + bb * sBb + hh * sBh;

    const int tid  = threadIdx.x;
    const int lane = tid & 31;
    const int w    = tid >> 5;        // 0..7
    const int wmW  = w & 1;           // wave row (x64)
    const int wnW  = w >> 1;          // wave col (x32)
    const long long bm = (long long)blockIdx.y * BM;
    const long long bn = (long long)blockIdx.x * BN;

    __shared__ __align__(16) bf16 As[2][BM * KPAD];
    __shared__ __align__(16) bf16 Bs[2][BN * KPAD];

    v8f acc[4][2];
#pragma unroll
    for (int t = 0; t < 4; ++t)
#pragma unroll
        for (int u = 0; u < 2; ++u)
#pragma unroll
            for (int i = 0; i < 8; ++i) acc[t][u][i] = 0.0f;

    const int mA  = lane & 15;
    const int kbA = (lane & 16) ? 8 : 0;    // A: lanes 16-31 hold K {8..15,24..31}
    const int nB  = lane & 15;
    const int kbB = (lane & 16) ? 8 : 0;    // B: lanes 16-31 hold K {8..15,24..31}

#if !USE_TDM
    const int c0row = tid >> 2;             // 4 x 8-elem chunks per 32-elem row
    const int c0off = (tid & 3) << 3;
    const int c1row = (tid + 256) >> 2;
    const int c1off = ((tid + 256) & 3) << 3;
#endif

    const int nt = K / BK;

    // ---- prologue: stage tile 0 into buffer 0 ----
#if USE_TDM
    if (w == 0) {
        tdm_load_2d(A + bm * lda, (unsigned)(unsigned long long)&As[0][0], BM, BK, lda);
        tdm_load_2d(B + bn * ldb, (unsigned)(unsigned long long)&Bs[0][0], BN, BK, ldb);
    }
#else
    async_copy16(A + (bm + c0row) * lda + c0off, &As[0][c0row * KPAD + c0off]);
    async_copy16(A + (bm + c1row) * lda + c1off, &As[0][c1row * KPAD + c1off]);
    async_copy16(B + (bn + c0row) * ldb + c0off, &Bs[0][c0row * KPAD + c0off]);
    async_copy16(B + (bn + c1row) * ldb + c1off, &Bs[0][c1row * KPAD + c1off]);
#endif

    for (int it = 0; it < nt; ++it) {
        const int cur = it & 1;
        const int k1  = (it + 1) * BK;

        // ---- stage tile it+1 into the other buffer, then fence tile it ----
#if USE_TDM
        if (w == 0) {
            if (it + 1 < nt) {
                tdm_load_2d(A + bm * lda + k1,
                            (unsigned)(unsigned long long)&As[cur ^ 1][0], BM, BK, lda);
                tdm_load_2d(B + bn * ldb + k1,
                            (unsigned)(unsigned long long)&Bs[cur ^ 1][0], BN, BK, ldb);
                wait_tensor<2>();   // TDM ops complete in order: tile `it` done
            } else {
                wait_tensor<0>();
            }
        }
#else
        if (it + 1 < nt) {
            async_copy16(A + (bm + c0row) * lda + k1 + c0off, &As[cur ^ 1][c0row * KPAD + c0off]);
            async_copy16(A + (bm + c1row) * lda + k1 + c1off, &As[cur ^ 1][c1row * KPAD + c1off]);
            async_copy16(B + (bn + c0row) * ldb + k1 + c0off, &Bs[cur ^ 1][c0row * KPAD + c0off]);
            async_copy16(B + (bn + c1row) * ldb + k1 + c1off, &Bs[cur ^ 1][c1row * KPAD + c1off]);
            wait_async<4>();        // async loads complete in order: tile `it` done
        } else {
            wait_async<0>();
        }
#endif
        if (it + 2 < nt) {          // warm tile it+2 (global_prefetch_b8)
            __builtin_prefetch(A + (bm + (tid >> 1)) * lda + k1 + BK, 0, 1);
            __builtin_prefetch(B + (bn + (tid >> 1)) * ldb + k1 + BK, 0, 1);
        }
        __syncthreads();            // buffer `cur` visible to all waves

        v16bf af[4], bfr[2];
#pragma unroll
        for (int t = 0; t < 4; ++t) {
            int r = (wmW * 64 + t * 16 + mA) * KPAD;
            af[t] = cat16(*reinterpret_cast<const v8bf*>(&As[cur][r + kbA]),
                          *reinterpret_cast<const v8bf*>(&As[cur][r + kbA + 16]));
        }
#pragma unroll
        for (int u = 0; u < 2; ++u) {
            int r = (wnW * 32 + u * 16 + nB) * KPAD;
            bfr[u] = cat16(*reinterpret_cast<const v8bf*>(&Bs[cur][r + kbB]),
                           *reinterpret_cast<const v8bf*>(&Bs[cur][r + kbB + 16]));
        }
#pragma unroll
        for (int t = 0; t < 4; ++t)
#pragma unroll
            for (int u = 0; u < 2; ++u)
                acc[t][u] = wmma_bf16(af[t], bfr[u], acc[t][u]);

        __syncthreads();            // all reads done before next overwrite
    }

    const int mBase = (lane & 16) ? 8 : 0;
    if (OUT_BF16) {
        bf16* C = (bf16*)Cg + bb * sCb + hh * sCh;
#pragma unroll
        for (int t = 0; t < 4; ++t)
#pragma unroll
            for (int u = 0; u < 2; ++u)
#pragma unroll
                for (int v = 0; v < 8; ++v) {
                    long long m = bm + wmW * 64 + t * 16 + mBase + v;
                    long long n = bn + wnW * 32 + u * 16 + nB;
                    C[m * ldc + n] = (bf16)acc[t][u][v];
                }
    } else {
        float* C = (float*)Cg + bb * sCb + hh * sCh;
#pragma unroll
        for (int t = 0; t < 4; ++t)
#pragma unroll
            for (int u = 0; u < 2; ++u)
#pragma unroll
                for (int v = 0; v < 8; ++v) {
                    long long m = bm + wmW * 64 + t * 16 + mBase + v;
                    long long n = bn + wnW * 32 + u * 16 + nB;
                    C[m * ldc + n] = acc[t][u][v];
                }
    }
}

// ---------------------------------------------------------------------------
__global__ __launch_bounds__(256) void cvt_bf16(
    const float* __restrict__ in, bf16* __restrict__ out)
{
    long long i = ((long long)blockIdx.x * 256 + threadIdx.x) * 8;
    float4 a = *reinterpret_cast<const float4*>(in + i);
    float4 b = *reinterpret_cast<const float4*>(in + i + 4);
    v8bf o;
    o[0] = (bf16)a.x; o[1] = (bf16)a.y; o[2] = (bf16)a.z; o[3] = (bf16)a.w;
    o[4] = (bf16)b.x; o[5] = (bf16)b.y; o[6] = (bf16)b.z; o[7] = (bf16)b.w;
    *reinterpret_cast<v8bf*>(out + i) = o;
}

template<typename T>
__global__ __launch_bounds__(256) void transpose_cvt(
    const T* __restrict__ in, bf16* __restrict__ out, int R, int C,
    long long sIn, long long sOut)
{
    const long long z = blockIdx.z;
    in  += z * sIn;
    out += z * sOut;
    __shared__ bf16 t[32][33];
    const int lr = threadIdx.x >> 5;
    const int lc = threadIdx.x & 31;
    const long long r0 = (long long)blockIdx.y * 32;
    const long long c0 = (long long)blockIdx.x * 32;
#pragma unroll
    for (int i = 0; i < 4; ++i) {
        int r = lr + 8 * i;
        t[lc][r] = (bf16)in[(r0 + r) * C + c0 + lc];
    }
    __syncthreads();
#pragma unroll
    for (int i = 0; i < 4; ++i) {
        int r = lr + 8 * i;
        out[(c0 + r) * R + r0 + lc] = t[r][lc];
    }
}

__global__ __launch_bounds__(256) void softmax_rows(
    float* __restrict__ attn, bf16* __restrict__ attnb, float scale)
{
    const long long row = blockIdx.x;
    float* p  = attn  + row * 2048ll;
    bf16*  pb = attnb + row * 2048ll;
    const int tid = threadIdx.x;
    __shared__ float red[8];

    float v[8];
    float mx = -__builtin_inff();
#pragma unroll
    for (int i = 0; i < 8; ++i) {
        v[i] = p[tid + (i << 8)] * scale;
        mx = fmaxf(mx, v[i]);
    }
#pragma unroll
    for (int off = 16; off > 0; off >>= 1) mx = fmaxf(mx, __shfl_xor(mx, off, 32));
    if ((tid & 31) == 0) red[tid >> 5] = mx;
    __syncthreads();
    mx = red[0];
#pragma unroll
    for (int i = 1; i < 8; ++i) mx = fmaxf(mx, red[i]);

    float sum = 0.0f;
#pragma unroll
    for (int i = 0; i < 8; ++i) {
        v[i] = __expf(v[i] - mx);
        sum += v[i];
    }
#pragma unroll
    for (int off = 16; off > 0; off >>= 1) sum += __shfl_xor(sum, off, 32);
    __syncthreads();
    if ((tid & 31) == 0) red[tid >> 5] = sum;
    __syncthreads();
    sum = 0.0f;
#pragma unroll
    for (int i = 0; i < 8; ++i) sum += red[i];

    float inv = 1.0f / sum;
#pragma unroll
    for (int i = 0; i < 8; ++i) {
        float r = v[i] * inv;
        p[tid + (i << 8)]  = r;
        pb[tid + (i << 8)] = (bf16)r;
    }
}

// ---------------------------------------------------------------------------
extern "C" void kernel_launch(void* const* d_in, const int* in_sizes, int n_in,
                              void* d_out, int out_size, void* d_ws, size_t ws_size,
                              hipStream_t stream)
{
    (void)in_sizes; (void)n_in; (void)out_size; (void)ws_size;

    constexpr int Bsz = 2, S = 2048, D = 2048, Hh = 16, DK = 128, L = 512;
    constexpr int MS  = Bsz * S;                        // 4096

    const float* query = (const float*)d_in[0];
    const float* key   = (const float*)d_in[1];
    const float* value = (const float*)d_in[2];
    const float* W_q   = (const float*)d_in[3];
    const float* W_kc  = (const float*)d_in[4];
    const float* W_vc  = (const float*)d_in[5];
    const float* W_ku  = (const float*)d_in[6];
    const float* W_vu  = (const float*)d_in[7];
    const float* W_o   = (const float*)d_in[8];

    float* out  = (float*)d_out;
    float* attn = out + (long long)Bsz * S * D;

    bf16* p    = (bf16*)d_ws;
    bf16* qb   = p; p += (long long)MS * D;
    bf16* kb   = p; p += (long long)MS * D;
    bf16* vb   = p; p += (long long)MS * D;
    bf16* WqT  = p; p += (long long)D * D;
    bf16* WkcT = p; p += (long long)D * L;
    bf16* WvcT = p; p += (long long)D * L;
    bf16* WkuT = p; p += (long long)L * D;
    bf16* WvuT = p; p += (long long)L * D;
    bf16* WoT  = p; p += (long long)D * D;
    bf16* Qm   = p; p += (long long)MS * D;
    bf16* Ck   = p; p += (long long)MS * L;
    bf16* Cv   = p; p += (long long)MS * L;
    bf16* Km   = p; p += (long long)MS * D;
    bf16* Vm   = p; p += (long long)MS * D;
    bf16* Vt   = p; p += (long long)MS * D;
    bf16* Ot   = p; p += (long long)MS * D;
    bf16* attnb = p;                                    // B*H*S*S

    dim3 blk(256);

    cvt_bf16<<<dim3((MS * (long long)D) / 2048), blk, 0, stream>>>(query, qb);
    cvt_bf16<<<dim3((MS * (long long)D) / 2048), blk, 0, stream>>>(key,   kb);
    cvt_bf16<<<dim3((MS * (long long)D) / 2048), blk, 0, stream>>>(value, vb);

    transpose_cvt<float><<<dim3(D / 32, D / 32, 1), blk, 0, stream>>>(W_q,  WqT,  D, D, 0, 0);
    transpose_cvt<float><<<dim3(L / 32, D / 32, 1), blk, 0, stream>>>(W_kc, WkcT, D, L, 0, 0);
    transpose_cvt<float><<<dim3(L / 32, D / 32, 1), blk, 0, stream>>>(W_vc, WvcT, D, L, 0, 0);
    transpose_cvt<float><<<dim3(D / 32, L / 32, 1), blk, 0, stream>>>(W_ku, WkuT, L, D, 0, 0);
    transpose_cvt<float><<<dim3(D / 32, L / 32, 1), blk, 0, stream>>>(W_vu, WvuT, L, D, 0, 0);
    transpose_cvt<float><<<dim3(D / 32, D / 32, 1), blk, 0, stream>>>(W_o,  WoT,  D, D, 0, 0);

    gemm_nt<true><<<dim3(D / BN, MS / BM, 1), blk, 0, stream>>>(
        qb, WqT, Qm, D, D, D, D, 0, 0, 0, 0, 0, 0, 1);
    gemm_nt<true><<<dim3(L / BN, MS / BM, 1), blk, 0, stream>>>(
        kb, WkcT, Ck, D, D, D, L, 0, 0, 0, 0, 0, 0, 1);
    gemm_nt<true><<<dim3(D / BN, MS / BM, 1), blk, 0, stream>>>(
        Ck, WkuT, Km, L, L, L, D, 0, 0, 0, 0, 0, 0, 1);
    gemm_nt<true><<<dim3(L / BN, MS / BM, 1), blk, 0, stream>>>(
        vb, WvcT, Cv, D, D, D, L, 0, 0, 0, 0, 0, 0, 1);
    gemm_nt<true><<<dim3(D / BN, MS / BM, 1), blk, 0, stream>>>(
        Cv, WvuT, Vm, L, L, L, D, 0, 0, 0, 0, 0, 0, 1);

    transpose_cvt<bf16><<<dim3(D / 32, S / 32, Bsz), blk, 0, stream>>>(
        Vm, Vt, S, D, (long long)S * D, (long long)D * S);

    const long long sQb = (long long)S * D, sQh = DK;
    const long long sAtb = (long long)Hh * S * S, sAth = (long long)S * S;
    gemm_nt<false><<<dim3(S / BN, S / BM, Bsz * Hh), blk, 0, stream>>>(
        Qm, Km, attn, DK, D, D, S,
        sQb, sQh, sQb, sQh, sAtb, sAth, Hh);

    softmax_rows<<<dim3(Bsz * Hh * S), blk, 0, stream>>>(
        attn, attnb, 0.08838834764831843f);

    gemm_nt<true><<<dim3(DK / BN, S / BM, Bsz * Hh), blk, 0, stream>>>(
        attnb, Vt, Ot, S, S, S, D,
        sAtb, sAth, (long long)D * S, (long long)DK * S, sQb, sQh, Hh);

    gemm_nt<false><<<dim3(D / BN, MS / BM, 1), blk, 0, stream>>>(
        Ot, WoT, out, D, D, D, D, 0, 0, 0, 0, 0, 0, 1);
}
